// Attention_81037442941378
// MI455X (gfx1250) — matrix operand order repaired and checked by reading, not compiled, not run on hardware
//
#include <hip/hip_runtime.h>

typedef __bf16 bf16_t;
typedef __bf16 v16bf __attribute__((ext_vector_type(16)));
typedef float  v8f   __attribute__((ext_vector_type(8)));
typedef unsigned int v4u __attribute__((ext_vector_type(4)));
typedef unsigned int v8u __attribute__((ext_vector_type(8)));

#define DIM   768
#define HEADS 12
#define HDIM  64
#define NBAT  2
#define SEQ   2048
#define NTOK  (NBAT * SEQ)   // 4096
#define QKVN  (3 * DIM)      // 2304

struct alignas(16) FragBits { int4 lo, hi; };

static __device__ __forceinline__ v16bf frag_ld(const bf16_t* p0, const bf16_t* p1) {
    FragBits f;
    f.lo = *reinterpret_cast<const int4*>(p0);
    f.hi = *reinterpret_cast<const int4*>(p1);
    return __builtin_bit_cast(v16bf, f);
}

static __device__ __forceinline__ v8f wmma_bf16(v16bf a, v16bf b, v8f c) {
    return __builtin_amdgcn_wmma_f32_16x16x32_bf16(false, a, false, b, (short)0, c, false, false);
}

// raw hardware exp2 (single v_exp_f32; args here are always <= 0)
static __device__ __forceinline__ float exp2_hw(float x) {
    return __builtin_amdgcn_exp2f(x);
}

// lane-xor max via ds_swizzle (group-of-32: and=0x1F, or=0, xor=MASK)
template<int XORMASK>
static __device__ __forceinline__ float fmax_xor(float x) {
    int s = __builtin_amdgcn_ds_swizzle(__float_as_int(x), (XORMASK << 10) | 0x1F);
    return fmaxf(x, __int_as_float(s));
}

// ---------------------------------------------------------------------------
// Tensor Data Mover: 2-D tile load Global -> LDS (D# groups 0 and 1 only).
// dims/tile in bf16 elements (data_size = 2B). LDS padding: pad_amount DWORD
// code after every 2^(pad_interval+1) DWORDs.
// ---------------------------------------------------------------------------
static __device__ __forceinline__ void tdm_load_2d(
    unsigned lds_addr, const void* gptr,
    unsigned dim0, unsigned dim1, unsigned long long stride0,
    unsigned pad_interval, unsigned pad_amount)
{
    unsigned long long ga = (unsigned long long)(size_t)gptr;
    v4u g0;
    g0[0] = 1u;                                      // count=1, user mode
    g0[1] = lds_addr;                                // LDS byte address
    g0[2] = (unsigned)ga;                            // global addr lo
    g0[3] = (unsigned)(ga >> 32) | 0x80000000u;      // addr hi | type=2
    v8u g1;
    g1[0] = (1u << 16)                               // data_size = 2 bytes
          | (1u << 20)                               // pad_enable
          | (pad_interval << 22) | (pad_amount << 25);
    g1[1] = (dim0 & 0xFFFFu) << 16;                  // tensor_dim0 lo (ab_addr=0)
    g1[2] = (dim0 >> 16) | ((dim1 & 0xFFFFu) << 16); // dim0 hi | dim1 lo
    g1[3] = (dim1 >> 16) | ((dim0 & 0xFFFFu) << 16); // dim1 hi | tile_dim0
    g1[4] = (dim1 & 0xFFFFu);                        // tile_dim1 (tile_dim2 = 0)
    g1[5] = (unsigned)stride0;                       // dim0 stride lo
    g1[6] = (unsigned)(stride0 >> 32) & 0xFFFFu;     // stride hi (dim1_stride=0)
    g1[7] = 0u;
    asm volatile("tensor_load_to_lds %0, %1" :: "s"(g0), "s"(g1) : "memory");
}

// ---------------------------------------------------------------------------
// f32 -> bf16 pack (vectorized x4)
// ---------------------------------------------------------------------------
__global__ __launch_bounds__(256) void cvt_bf16_kernel(
    const float* __restrict__ in, bf16_t* __restrict__ out, int n4)
{
    int i = blockIdx.x * 256 + threadIdx.x;
    if (i < n4) {
        float4 v = reinterpret_cast<const float4*>(in)[i];
        union { bf16_t h[4]; int2 b; } pk;
        pk.h[0] = (bf16_t)v.x; pk.h[1] = (bf16_t)v.y;
        pk.h[2] = (bf16_t)v.z; pk.h[3] = (bf16_t)v.w;
        reinterpret_cast<int2*>(out)[i] = pk.b;
    }
}

// ---------------------------------------------------------------------------
// NT GEMM: C[M,N] = A[M,K] * W[N,K]^T (+bias). bf16 in, f32 out.
// Block 256 (8 waves), block tile 128x128, K-step 64, TDM double-buffered.
// Wave tile 32(M) x 64(N): 4 M-groups x 2 N-groups; 16 wmma / wave / k-step.
// ---------------------------------------------------------------------------
template<bool ADD_BIAS>
__global__ __launch_bounds__(256) void gemm_nt_kernel(
    const bf16_t* __restrict__ A, const bf16_t* __restrict__ W,
    const float* __restrict__ bias, float* __restrict__ C,
    int M, int N, int K)
{
    (void)M;
    __shared__ bf16_t sA[2][128][72];   // 64 + 8 pad per row
    __shared__ bf16_t sB[2][128][72];   // B column-major (= W rows)

    const int tid = threadIdx.x, lane = tid & 31, wave = tid >> 5;
    const int mg = wave & 3, ng = wave >> 2;
    const int m0 = blockIdx.y * 128, n0 = blockIdx.x * 128;
    const int l15 = lane & 15, h8 = (lane >> 4) * 8, h16 = (lane >> 4) * 16;

    v8f acc[2][4];
#pragma unroll
    for (int mi = 0; mi < 2; ++mi)
#pragma unroll
        for (int n = 0; n < 4; ++n)
            acc[mi][n] = (v8f){0.f,0.f,0.f,0.f,0.f,0.f,0.f,0.f};

    const int nsteps = K / 64;
    if (wave == 0) {
        tdm_load_2d((unsigned)(size_t)&sA[0][0][0], A + (size_t)m0 * K,
                    64, 128, (unsigned long long)K, 4, 3);
        tdm_load_2d((unsigned)(size_t)&sB[0][0][0], W + (size_t)n0 * K,
                    64, 128, (unsigned long long)K, 4, 3);
    }
    for (int ks = 0; ks < nsteps; ++ks) {
        const int cur = ks & 1;
        if (wave == 0) {
            if (ks + 1 < nsteps) {
                const int k0n = (ks + 1) * 64;
                tdm_load_2d((unsigned)(size_t)&sA[cur ^ 1][0][0],
                            A + (size_t)m0 * K + k0n, 64, 128, (unsigned long long)K, 4, 3);
                tdm_load_2d((unsigned)(size_t)&sB[cur ^ 1][0][0],
                            W + (size_t)n0 * K + k0n, 64, 128, (unsigned long long)K, 4, 3);
                __builtin_amdgcn_s_wait_tensorcnt(2);   // current buffer done
            } else {
                __builtin_amdgcn_s_wait_tensorcnt(0);
            }
        }
        __syncthreads();

#pragma unroll
        for (int kk = 0; kk < 2; ++kk) {
            const bf16_t* ap0 = &sA[cur][mg * 32 + l15][kk * 32 + h8];
            const bf16_t* ap1 = &sA[cur][mg * 32 + 16 + l15][kk * 32 + h8];
            v16bf a0 = frag_ld(ap0, ap0 + 16);
            v16bf a1 = frag_ld(ap1, ap1 + 16);
#pragma unroll
            for (int n = 0; n < 4; ++n) {
                const bf16_t* bp = &sB[cur][ng * 64 + n * 16 + l15][kk * 32 + h8];
                v16bf b = frag_ld(bp, bp + 16);
                acc[0][n] = wmma_bf16(a0, b, acc[0][n]);
                acc[1][n] = wmma_bf16(a1, b, acc[1][n]);
            }
        }
        __syncthreads();
    }

    const int rbase = (lane >> 4) * 8;
#pragma unroll
    for (int mi = 0; mi < 2; ++mi)
#pragma unroll
    for (int n = 0; n < 4; ++n) {
        int col = n0 + ng * 64 + n * 16 + l15;
        float bv = 0.f;
        if constexpr (ADD_BIAS) bv = bias[col];
#pragma unroll
        for (int r = 0; r < 8; ++r) {
            int row = m0 + mg * 32 + mi * 16 + rbase + r;
            C[(size_t)row * N + col] = acc[mi][n][r] + bv;
        }
    }
}

// ---------------------------------------------------------------------------
// RoPE on Q,K + bf16 pack: qb/kb [B,H,S,D]. Q is pre-scaled by
// HDIM^-0.5 * log2(e) so flash scores are in the log2 domain (exp2 softmax).
// vtb [B,H,D,S] (V transposed). One thread per (b,h,s,d-pair).
// ---------------------------------------------------------------------------
__global__ __launch_bounds__(256) void rope_pack_kernel(
    const float* __restrict__ qkv, bf16_t* __restrict__ qb,
    bf16_t* __restrict__ kb, bf16_t* __restrict__ vtb)
{
    int idx = blockIdx.x * 256 + threadIdx.x;
    int dp  = idx & 31;
    int s   = (idx >> 5) & (SEQ - 1);
    int bh  = idx >> 16;                 // SEQ*32 = 65536 per (b,h)
    int d   = dp * 2;
    int b   = bh / HEADS, h = bh % HEADS;

    size_t src = (size_t)(b * SEQ + s) * QKVN + (size_t)h * HDIM + d;
    float q0 = qkv[src],            q1 = qkv[src + 1];
    float k0 = qkv[src + DIM],      k1 = qkv[src + DIM + 1];
    float v0 = qkv[src + 2 * DIM],  v1 = qkv[src + 2 * DIM + 1];

    float ang = (float)s * __expf(-(float)dp * 0.2878231364f); // ln(1e4)/32
    float sn, cs;
    sincosf(ang, &sn, &cs);

    const float QS = 0.125f * 1.4426950408889634f;   // HDIM^-0.5 * log2(e)
    float qe = (q0 * cs - q1 * sn) * QS;
    float qo = (q1 * cs + q0 * sn) * QS;
    float ke = k0 * cs - k1 * sn, ko = k1 * cs + k0 * sn;

    size_t dst = (size_t)bh * SEQ * HDIM + (size_t)s * HDIM + d;
    qb[dst] = (bf16_t)qe; qb[dst + 1] = (bf16_t)qo;
    kb[dst] = (bf16_t)ke; kb[dst + 1] = (bf16_t)ko;

    size_t dv = (size_t)bh * HDIM * SEQ + (size_t)d * SEQ + s;
    vtb[dv] = (bf16_t)v0; vtb[dv + SEQ] = (bf16_t)v1;
}

// ---------------------------------------------------------------------------
// Flash attention: grid (S/128, B*H), block 256 (8 waves x 16 q-rows).
// TDM double-buffered K/V^T tiles; scores in log2 domain (hardware exp2);
// row-max via ds_swizzle xor butterflies; row-sum via WMMA against ones.
// ---------------------------------------------------------------------------
__global__ __launch_bounds__(256) void flash_attn_kernel(
    const bf16_t* __restrict__ qb, const bf16_t* __restrict__ kb,
    const bf16_t* __restrict__ vtb, bf16_t* __restrict__ aob)
{
    __shared__ bf16_t sK[2][64][72];     // [key][d], 64 + 8 pad
    __shared__ bf16_t sV[2][64][72];     // [d][key]
    __shared__ bf16_t sP[8][16][72];     // per-wave probs 16 x 64

    const int tid  = threadIdx.x, lane = tid & 31, wave = tid >> 5;
    const int l15  = lane & 15, h8 = (lane >> 4) * 8, h16 = (lane >> 4) * 16;
    const int bh = blockIdx.y, q0 = blockIdx.x * 128;

    const bf16_t* qrow = qb + (size_t)bh * SEQ * HDIM + (size_t)(q0 + wave * 16 + l15) * HDIM;
    v16bf qa0 = frag_ld(qrow + h8,      qrow + 16 + h8);
    v16bf qa1 = frag_ld(qrow + 32 + h8, qrow + 48 + h8);

    v16bf ones;
#pragma unroll
    for (int i = 0; i < 16; ++i) ones[i] = (bf16_t)1.0f;

    v8f ao[4];
    float mrow[8], lrow[8];
#pragma unroll
    for (int n = 0; n < 4; ++n) ao[n] = (v8f){0.f,0.f,0.f,0.f,0.f,0.f,0.f,0.f};
#pragma unroll
    for (int r = 0; r < 8; ++r) { mrow[r] = -3.0e38f; lrow[r] = 0.f; }

    const bf16_t* kg = kb  + (size_t)bh * SEQ * HDIM;
    const bf16_t* vg = vtb + (size_t)bh * HDIM * SEQ;

    if (wave == 0) {
        tdm_load_2d((unsigned)(size_t)&sK[0][0][0], kg, 64, 64, 64ull, 4, 3);
        tdm_load_2d((unsigned)(size_t)&sV[0][0][0], vg, 64, 64, (unsigned long long)SEQ, 4, 3);
    }
    for (int kt = 0; kt < SEQ / 64; ++kt) {
        const int cur = kt & 1;
        if (wave == 0) {
            if (kt + 1 < SEQ / 64) {
                tdm_load_2d((unsigned)(size_t)&sK[cur ^ 1][0][0],
                            kg + (size_t)(kt + 1) * 64 * HDIM, 64, 64, 64ull, 4, 3);
                tdm_load_2d((unsigned)(size_t)&sV[cur ^ 1][0][0],
                            vg + (size_t)(kt + 1) * 64, 64, 64, (unsigned long long)SEQ, 4, 3);
                __builtin_amdgcn_s_wait_tensorcnt(2);
            } else {
                __builtin_amdgcn_s_wait_tensorcnt(0);
            }
        }
        __syncthreads();

        // scores (log2 domain) = Qs K^T
        v8f sc[4];
#pragma unroll
        for (int n = 0; n < 4; ++n) {
            const bf16_t* b0 = &sK[cur][n * 16 + l15][h8];
            const bf16_t* b1 = &sK[cur][n * 16 + l15][32 + h8];
            v8f cf = (v8f){0.f,0.f,0.f,0.f,0.f,0.f,0.f,0.f};
            cf = wmma_bf16(qa0, frag_ld(b0, b0 + 16), cf);
            cf = wmma_bf16(qa1, frag_ld(b1, b1 + 16), cf);
            sc[n] = cf;
        }

        // row max: frag-max then 16-lane xor butterfly via ds_swizzle
        float tm[8];
#pragma unroll
        for (int r = 0; r < 8; ++r)
            tm[r] = fmaxf(fmaxf(sc[0][r], sc[1][r]), fmaxf(sc[2][r], sc[3][r]));
#pragma unroll
        for (int r = 0; r < 8; ++r) tm[r] = fmax_xor<8>(tm[r]);
#pragma unroll
        for (int r = 0; r < 8; ++r) tm[r] = fmax_xor<4>(tm[r]);
#pragma unroll
        for (int r = 0; r < 8; ++r) tm[r] = fmax_xor<2>(tm[r]);
#pragma unroll
        for (int r = 0; r < 8; ++r) tm[r] = fmax_xor<1>(tm[r]);

        float alpha[8];
#pragma unroll
        for (int r = 0; r < 8; ++r) {
            float mnew = fmaxf(mrow[r], tm[r]);
            alpha[r] = exp2_hw(mrow[r] - mnew);    // single v_exp_f32
            mrow[r] = mnew;
        }

        // P = exp2(S - m), stored bf16 to per-wave LDS (C-layout -> A-layout)
#pragma unroll
        for (int n = 0; n < 4; ++n)
#pragma unroll
            for (int r = 0; r < 8; ++r)
                sP[wave][h8 + r][n * 16 + l15] = (bf16_t)exp2_hw(sc[n][r] - mrow[r]);
        asm volatile("s_wait_dscnt 0" ::: "memory");

        const bf16_t* pp = &sP[wave][l15][h8];
        v16bf pa0 = frag_ld(pp,      pp + 16);   // keys 0..31
        v16bf pa1 = frag_ld(pp + 32, pp + 48);   // keys 32..63

        // row sums via WMMA against ones (every column of D = row sum)
        v8f sacc = (v8f){0.f,0.f,0.f,0.f,0.f,0.f,0.f,0.f};
        sacc = wmma_bf16(pa0, ones, sacc);
        sacc = wmma_bf16(pa1, ones, sacc);
#pragma unroll
        for (int r = 0; r < 8; ++r) lrow[r] = lrow[r] * alpha[r] + sacc[r];

        // rescale O and accumulate P V
#pragma unroll
        for (int n = 0; n < 4; ++n) {
#pragma unroll
            for (int r = 0; r < 8; ++r) ao[n][r] *= alpha[r];
        }
#pragma unroll
        for (int n = 0; n < 4; ++n) {
            const bf16_t* v0 = &sV[cur][n * 16 + l15][h8];
            const bf16_t* v1 = &sV[cur][n * 16 + l15][32 + h8];
            ao[n] = wmma_bf16(pa0, frag_ld(v0, v0 + 16), ao[n]);
            ao[n] = wmma_bf16(pa1, frag_ld(v1, v1 + 16), ao[n]);
        }
        __syncthreads();
    }

    // write O / l as bf16 to [B, S, C] for the projection GEMM
    const int b = bh / HEADS, h = bh % HEADS, rbase = h8;
#pragma unroll
    for (int n = 0; n < 4; ++n) {
        int col = h * HDIM + n * 16 + l15;
#pragma unroll
        for (int r = 0; r < 8; ++r) {
            int s = q0 + wave * 16 + rbase + r;
            aob[(size_t)(b * SEQ + s) * DIM + col] = (bf16_t)(ao[n][r] / lrow[r]);
        }
    }
}

// ---------------------------------------------------------------------------
extern "C" void kernel_launch(void* const* d_in, const int* in_sizes, int n_in,
                              void* d_out, int out_size, void* d_ws, size_t ws_size,
                              hipStream_t stream) {
    (void)in_sizes; (void)n_in; (void)out_size; (void)ws_size;
    const float* x      = (const float*)d_in[0];
    const float* W_qkv  = (const float*)d_in[1];
    const float* W_proj = (const float*)d_in[2];
    const float* b_proj = (const float*)d_in[3];
    float* out = (float*)d_out;

    char* w = (char*)d_ws;
    float*  qkv  = (float*)w;   w += (size_t)NTOK * QKVN * sizeof(float);
    bf16_t* qb   = (bf16_t*)w;  w += (size_t)NTOK * DIM * sizeof(bf16_t);
    bf16_t* kbuf = (bf16_t*)w;  w += (size_t)NTOK * DIM * sizeof(bf16_t);
    bf16_t* vtb  = (bf16_t*)w;  w += (size_t)NTOK * DIM * sizeof(bf16_t);
    bf16_t* xb   = (bf16_t*)w;  w += (size_t)NTOK * DIM * sizeof(bf16_t);
    bf16_t* wqb  = (bf16_t*)w;  w += (size_t)QKVN * DIM * sizeof(bf16_t);
    bf16_t* wpb  = (bf16_t*)w;  w += (size_t)DIM * DIM * sizeof(bf16_t);
    bf16_t* aob  = (bf16_t*)w;  // NTOK * DIM bf16

    // 0) pre-convert activations & weights to bf16 (read-once)
    cvt_bf16_kernel<<<(NTOK * DIM / 4) / 256, 256, 0, stream>>>(x, xb, NTOK * DIM / 4);
    cvt_bf16_kernel<<<(QKVN * DIM / 4) / 256, 256, 0, stream>>>(W_qkv, wqb, QKVN * DIM / 4);
    cvt_bf16_kernel<<<(DIM * DIM / 4) / 256, 256, 0, stream>>>(W_proj, wpb, DIM * DIM / 4);
    // 1) QKV projection: [4096,768] x [2304,768]^T -> f32
    gemm_nt_kernel<false><<<dim3(QKVN / 128, NTOK / 128), 256, 0, stream>>>(
        xb, wqb, nullptr, qkv, NTOK, QKVN, DIM);
    // 2) RoPE + pack (Q scaled, V transposed)
    rope_pack_kernel<<<(NTOK * HEADS * 32) / 256, 256, 0, stream>>>(
        qkv, qb, kbuf, vtb);
    // 3) Flash attention -> bf16 [B,S,C]
    flash_attn_kernel<<<dim3(SEQ / 128, NBAT * HEADS), 256, 0, stream>>>(
        qb, kbuf, vtb, aob);
    // 4) Output projection + bias -> f32 out
    gemm_nt_kernel<true><<<dim3(DIM / 128, NTOK / 128), 256, 0, stream>>>(
        aob, wpb, b_proj, out, NTOK, DIM, DIM);
}
